// CrossAttention_44504451121360
// MI455X (gfx1250) — matrix operand order repaired and checked by reading, not compiled, not run on hardware
//
#include <hip/hip_runtime.h>

typedef _Float16 half_t;
typedef __attribute__((ext_vector_type(16))) _Float16 v16h;
typedef __attribute__((ext_vector_type(8)))  _Float16 v8h;
typedef __attribute__((ext_vector_type(8)))  float    v8f;

#define B_  4
#define LQ_ 2048
#define LC_ 2048
#define D_  1024
#define H_  16
#define HD_ 64

union AF { v16h v; v8h h[2]; };

typedef __attribute__((address_space(3))) char lds_char_t;
__device__ __forceinline__ unsigned ldsAddr(void* p) {
    return (unsigned)(unsigned long long)(lds_char_t*)p;
}

// ---------------------------------------------------------------------------
// fp32 -> fp16 conversion (vectorized, 8 elems/thread/iter)
// ---------------------------------------------------------------------------
__global__ void f32_to_f16(const float* __restrict__ src, half_t* __restrict__ dst, int n) {
    int tid    = blockIdx.x * blockDim.x + threadIdx.x;
    int stride = gridDim.x * blockDim.x;
    for (int base = tid * 8; base < n; base += stride * 8) {
        v8h h;
#pragma unroll
        for (int e = 0; e < 8; ++e) h[e] = (half_t)src[base + e];
        *(v8h*)(dst + base) = h;
    }
}

// ---------------------------------------------------------------------------
// WMMA GEMM:  C = A(MxK, f16 row-major) * W^T (W is NxK f16 row-major) + bias
// Wave computes a 32x64 tile (2 A-fragments x 4 B-fragments, 8 accumulators).
// MODE 0: f16 out, (B,H,L,HD) layout        (Q, K)
// MODE 1: f16 out, (B,H,HD,L) layout (V^T)  (V)
// MODE 2: f32 out, row-major MxN            (final projection -> d_out)
// ---------------------------------------------------------------------------
template<int MODE>
__global__ void __launch_bounds__(128)
gemm16(const half_t* __restrict__ A, const half_t* __restrict__ W,
       const float* __restrict__ bias, void* __restrict__ Out,
       int M, int N, int K, int L)
{
    const int lane    = threadIdx.x & 31;
    const int wave    = threadIdx.x >> 5;
    const int m0      = blockIdx.x * 128 + wave * 32;
    const int n0      = blockIdx.y * 64;
    const int nlane   = lane & 15;
    const int half_id = lane >> 4;
    const int koffA   = half_id * 16;
    const int koffB   = half_id * 16;

    const half_t* arow0 = A + (size_t)(m0 + nlane) * K;
    const half_t* arow1 = A + (size_t)(m0 + 16 + nlane) * K;
    const half_t* wbase = W + (size_t)(n0 + nlane) * K;

    v8f acc[2][4] = {};

    for (int k0 = 0; k0 < K; k0 += 32) {
        if (k0 + 64 < K) {   // global_prefetch_b8 of next-next k-slab
            __builtin_prefetch(arow0 + k0 + 64, 0, 3);
            __builtin_prefetch(wbase + k0 + 64, 0, 3);
        }
        AF a0, a1;
        a0.h[0] = *(const v8h*)(arow0 + k0 + koffA);
        a0.h[1] = *(const v8h*)(arow0 + k0 + koffA + 8);
        a1.h[0] = *(const v8h*)(arow1 + k0 + koffA);
        a1.h[1] = *(const v8h*)(arow1 + k0 + koffA + 8);
#pragma unroll
        for (int j = 0; j < 4; ++j) {
            const half_t* wrow = W + (size_t)(n0 + j * 16 + nlane) * K + k0 + koffB;
            v16h bf = *(const v16h*)wrow;
            acc[0][j] = __builtin_amdgcn_wmma_f32_16x16x32_f16(
                            false, a0.v, false, bf, (short)0, acc[0][j], false, false);
            acc[1][j] = __builtin_amdgcn_wmma_f32_16x16x32_f16(
                            false, a1.v, false, bf, (short)0, acc[1][j], false, false);
        }
    }

#pragma unroll
    for (int g = 0; g < 2; ++g) {
#pragma unroll
        for (int j = 0; j < 4; ++j) {
            const int   n  = n0 + j * 16 + nlane;
            const float bj = bias[n];
#pragma unroll
            for (int r = 0; r < 8; ++r) {
                const int   m   = m0 + g * 16 + r + half_id * 8;
                const float val = acc[g][j][r] + bj;
                if (MODE == 2) {
                    ((float*)Out)[(size_t)m * N + n] = val;
                } else {
                    const int b  = m / L, l = m % L;
                    const int h  = n >> 6, hd = n & 63;
                    const half_t hv = (half_t)val;
                    if (MODE == 0)
                        ((half_t*)Out)[(((size_t)(b * H_ + h) * L + l) << 6) + hd] = hv;
                    else
                        ((half_t*)Out)[((size_t)(b * H_ + h) * HD_ + hd) * (size_t)L + l] = hv;
                }
            }
        }
    }
}

// ---------------------------------------------------------------------------
// Async staging of one 64-key K/V chunk into LDS (per-thread 4+4 B128 moves).
// K chunk: contiguous 8KB (keys are rows of 128B). V^T chunk: 64 rows x 128B,
// row stride LC*2 bytes.
// ---------------------------------------------------------------------------
__device__ __forceinline__ void stage_async(unsigned kLds, unsigned vLds,
                                            unsigned long long kG, unsigned long long vG,
                                            int tid)
{
#pragma unroll
    for (int i = 0; i < 4; ++i) {
        const unsigned idx  = (unsigned)(i * 128 + tid);
        const unsigned off  = idx * 16;
        asm volatile("global_load_async_to_lds_b128 %0, %1, %2"
                     :: "v"(kLds + off), "v"(off), "s"(kG) : "memory");
        const unsigned goff = (idx >> 3) * (LC_ * 2) + (idx & 7) * 16;
        asm volatile("global_load_async_to_lds_b128 %0, %1, %2"
                     :: "v"(vLds + off), "v"(goff), "s"(vG) : "memory");
    }
}

// ---------------------------------------------------------------------------
// Flash attention: wave owns 16 queries, block streams 64 keys/iter through
// double-buffered async-LDS K/V tiles shared by all 4 waves.
// LDS map (bytes): [0,8K) kBuf0 | [8K,16K) kBuf1 | [16K,24K) vBuf0 |
//                  [24K,32K) vBuf1 | [32K, 32K+9216) per-wave P tiles
// ---------------------------------------------------------------------------
__global__ void __launch_bounds__(128)
attn_fwd(const half_t* __restrict__ Q, const half_t* __restrict__ Kh,
         const half_t* __restrict__ Vt, half_t* __restrict__ Oc)
{
    __shared__ __align__(16) unsigned char smem[41984];
    const int tid     = threadIdx.x;
    const int lane    = tid & 31;
    const int wave    = tid >> 5;
    const int nQB     = LQ_ / 64;
    const int bh      = blockIdx.x / nQB;   // b*H + h
    const int qb      = blockIdx.x % nQB;
    const int q0      = qb * 64 + wave * 16;
    const int b       = bh / H_;
    const int h       = bh % H_;
    const int nlane   = lane & 15;
    const int half_id = lane >> 4;
    const int koffA   = half_id * 16;
    const int koffB   = half_id * 16;

    const unsigned sBase = ldsAddr(smem);           // LDS byte address of smem
    half_t* pT = (half_t*)(smem + 32768) + wave * (16 * 72);   // 144B pitch

    const half_t* kbase = Kh + ((size_t)bh * LC_ << 6);
    const half_t* vbase = Vt + (size_t)bh * HD_ * LC_;
    const unsigned long long kgb = (unsigned long long)kbase;
    const unsigned long long vgb = (unsigned long long)vbase;

    // Q fragments, pre-scaled by (1/sqrt(64)) * log2(e)
    const float qscale = 0.125f * 1.44269504088896340736f;
    const half_t* qrow = Q + (((size_t)bh * LQ_ + q0 + nlane) << 6);
    AF aq0, aq1;
    aq0.h[0] = *(const v8h*)(qrow + koffA);
    aq0.h[1] = *(const v8h*)(qrow + koffA + 8);
    aq1.h[0] = *(const v8h*)(qrow + 32 + koffA);
    aq1.h[1] = *(const v8h*)(qrow + 32 + koffA + 8);
#pragma unroll
    for (int e = 0; e < 16; ++e) {
        aq0.v[e] = (half_t)((float)aq0.v[e] * qscale);
        aq1.v[e] = (half_t)((float)aq1.v[e] * qscale);
    }

    v8f o[4] = {};
    float mrun[8], lrun[8];
#pragma unroll
    for (int r = 0; r < 8; ++r) { mrun[r] = -1e30f; lrun[r] = 0.f; }

    // prologue: stage chunk 0 into buffer 0
    stage_async(sBase + 0u, sBase + 16384u, kgb, vgb, tid);
    asm volatile("s_wait_asynccnt 0" ::: "memory");
    __syncthreads();

    for (int kc = 0; kc < LC_; kc += 64) {
        const int cur = (kc >> 6) & 1;
        if (kc + 64 < LC_)   // stage next chunk into the other buffer
            stage_async(sBase + (cur ? 0u : 8192u),
                        sBase + (cur ? 16384u : 24576u),
                        kgb + (unsigned long long)(kc + 64) * 128,
                        vgb + (unsigned long long)(kc + 64) * 2, tid);
        half_t* kTile = (half_t*)(smem + (cur ? 8192u : 0u));
        half_t* vTile = (half_t*)(smem + (cur ? 24576u : 16384u));

        // ---- S = Q K^T : four 16(q) x 16(key) tiles
        v8f s[4];
#pragma unroll
        for (int t = 0; t < 4; ++t) {
            const half_t* kp = kTile + (t * 16 + nlane) * 64 + koffB;
            v16h blo = *(const v16h*)kp;          // hd 0..31 column
            v16h bhi = *(const v16h*)(kp + 32);   // hd 32..63 column
            v8f a = {};
            a = __builtin_amdgcn_wmma_f32_16x16x32_f16(false, aq0.v, false, blo, (short)0, a, false, false);
            a = __builtin_amdgcn_wmma_f32_16x16x32_f16(false, aq1.v, false, bhi, (short)0, a, false, false);
            s[t] = a;
        }

        // ---- online softmax (row = C-register r, reduced over 16 lanes)
#pragma unroll
        for (int r = 0; r < 8; ++r) {
            float t = fmaxf(fmaxf(s[0][r], s[1][r]), fmaxf(s[2][r], s[3][r]));
            t = fmaxf(t, __shfl_xor(t, 1, 32));
            t = fmaxf(t, __shfl_xor(t, 2, 32));
            t = fmaxf(t, __shfl_xor(t, 4, 32));
            t = fmaxf(t, __shfl_xor(t, 8, 32));
            const float nm   = fmaxf(mrun[r], t);
            const float corr = exp2f(mrun[r] - nm);
            mrun[r] = nm;
            const float e0 = exp2f(s[0][r] - nm);
            const float e1 = exp2f(s[1][r] - nm);
            const float e2 = exp2f(s[2][r] - nm);
            const float e3 = exp2f(s[3][r] - nm);
            float ps = (e0 + e1) + (e2 + e3);
            ps += __shfl_xor(ps, 1, 32);
            ps += __shfl_xor(ps, 2, 32);
            ps += __shfl_xor(ps, 4, 32);
            ps += __shfl_xor(ps, 8, 32);
            lrun[r] = lrun[r] * corr + ps;
#pragma unroll
            for (int j = 0; j < 4; ++j) o[j][r] *= corr;
            const int row = r + half_id * 8;
            pT[row * 72 + nlane]      = (half_t)e0;
            pT[row * 72 + 16 + nlane] = (half_t)e1;
            pT[row * 72 + 32 + nlane] = (half_t)e2;
            pT[row * 72 + 48 + nlane] = (half_t)e3;
        }
        asm volatile("s_wait_dscnt 0" ::: "memory");

        // ---- P in A-layout (two 16x32 fragments)
        AF pa0, pa1;
        pa0.h[0] = *(const v8h*)(pT + nlane * 72 + koffA);
        pa0.h[1] = *(const v8h*)(pT + nlane * 72 + koffA + 8);
        pa1.h[0] = *(const v8h*)(pT + nlane * 72 + 32 + koffA);
        pa1.h[1] = *(const v8h*)(pT + nlane * 72 + 32 + koffA + 8);

        // ---- O += P * V : 4 hd-fragments x 2 key-steps
#pragma unroll
        for (int j = 0; j < 4; ++j) {
            const half_t* vp = vTile + (j * 16 + nlane) * 64 + koffB;
            v16h bv0 = *(const v16h*)vp;          // keys 0..31 of chunk
            v16h bv1 = *(const v16h*)(vp + 32);   // keys 32..63 of chunk
            o[j] = __builtin_amdgcn_wmma_f32_16x16x32_f16(false, pa0.v, false, bv0, (short)0, o[j], false, false);
            o[j] = __builtin_amdgcn_wmma_f32_16x16x32_f16(false, pa1.v, false, bv1, (short)0, o[j], false, false);
        }

        asm volatile("s_wait_asynccnt 0" ::: "memory");  // own next-chunk loads landed
        __syncthreads();                                  // safe to overwrite old buffer
    }

    // ---- normalize and write head-concat layout (B, LQ, D)
#pragma unroll
    for (int r = 0; r < 8; ++r) {
        const float inv = 1.0f / lrun[r];
        const int   q   = q0 + r + half_id * 8;
        const size_t ro = (size_t)(b * LQ_ + q) * D_ + h * HD_ + nlane;
        Oc[ro +  0] = (half_t)(o[0][r] * inv);
        Oc[ro + 16] = (half_t)(o[1][r] * inv);
        Oc[ro + 32] = (half_t)(o[2][r] * inv);
        Oc[ro + 48] = (half_t)(o[3][r] * inv);
    }
}

// ---------------------------------------------------------------------------
extern "C" void kernel_launch(void* const* d_in, const int* in_sizes, int n_in,
                              void* d_out, int out_size, void* d_ws, size_t ws_size,
                              hipStream_t stream)
{
    const float* x_cond = (const float*)d_in[0];
    const float* x      = (const float*)d_in[1];
    const float* wq = (const float*)d_in[2];
    const float* bq = (const float*)d_in[3];
    const float* wk = (const float*)d_in[4];
    const float* bk = (const float*)d_in[5];
    const float* wv = (const float*)d_in[6];
    const float* bv = (const float*)d_in[7];
    const float* wo = (const float*)d_in[8];
    const float* bo = (const float*)d_in[9];
    float* out = (float*)d_out;

    const size_t NX = (size_t)B_ * LQ_ * D_;   // 8,388,608
    const size_t NW = (size_t)D_ * D_;         // 1,048,576
    half_t* ws  = (half_t*)d_ws;
    half_t* xh  = ws;  ws += NX;
    half_t* xch = ws;  ws += NX;
    half_t* wqh = ws;  ws += NW;
    half_t* wkh = ws;  ws += NW;
    half_t* wvh = ws;  ws += NW;
    half_t* woh = ws;  ws += NW;
    half_t* Qh  = ws;  ws += NX;
    half_t* Kh  = ws;  ws += NX;
    half_t* Vt  = ws;  ws += NX;
    half_t* Oc  = ws;  ws += NX;

    f32_to_f16<<<4096, 256, 0, stream>>>(x,      xh,  (int)NX);
    f32_to_f16<<<4096, 256, 0, stream>>>(x_cond, xch, (int)NX);
    f32_to_f16<<<512,  256, 0, stream>>>(wq, wqh, (int)NW);
    f32_to_f16<<<512,  256, 0, stream>>>(wk, wkh, (int)NW);
    f32_to_f16<<<512,  256, 0, stream>>>(wv, wvh, (int)NW);
    f32_to_f16<<<512,  256, 0, stream>>>(wo, woh, (int)NW);

    dim3 gg((B_ * LQ_) / 128, D_ / 64);   // 64 x 16 blocks, 128 threads (4 waves)
    gemm16<0><<<gg, 128, 0, stream>>>(xh,  wqh, bq, Qh, B_ * LQ_, D_, D_, LQ_);
    gemm16<0><<<gg, 128, 0, stream>>>(xch, wkh, bk, Kh, B_ * LC_, D_, D_, LC_);
    gemm16<1><<<gg, 128, 0, stream>>>(xch, wvh, bv, Vt, B_ * LC_, D_, D_, LC_);

    attn_fwd<<<B_ * H_ * (LQ_ / 64), 128, 0, stream>>>(Qh, Kh, Vt, Oc);

    gemm16<2><<<gg, 128, 0, stream>>>(Oc, woh, bo, out, B_ * LQ_, D_, D_, LQ_);
}
